// Retnet_66460323938438
// MI455X (gfx1250) — matrix operand order repaired and checked by reading, not compiled, not run on hardware
//
#include <hip/hip_runtime.h>
#include <hip/hip_bf16.h>

typedef __bf16 bh16;
typedef __attribute__((ext_vector_type(16))) __bf16 v16bf;
typedef __attribute__((ext_vector_type(8)))  float  v8f;
typedef __attribute__((ext_vector_type(4)))  int    v4i;

// ---------------------------------------------------------------- async copy
#if __has_builtin(__builtin_amdgcn_global_load_async_to_lds_b128)
#define HAVE_ASYNC_LDS 1
__device__ __forceinline__ void async_copy_b128(void* lds_dst, const void* gsrc) {
  // (global_src, lds_dst, imm_offset, cpol) with clang ext-vector int4 pointers
  __builtin_amdgcn_global_load_async_to_lds_b128(
      (v4i*)(uintptr_t)gsrc, (v4i*)lds_dst, 0, 0);
}
__device__ __forceinline__ void wait_async0() {
#if __has_builtin(__builtin_amdgcn_s_wait_asynccnt)
  __builtin_amdgcn_s_wait_asynccnt(0);
#else
  asm volatile("s_wait_asynccnt 0x0" ::: "memory");
#endif
}
#else
#define HAVE_ASYNC_LDS 0
#endif

// ---------------------------------------------------------------- WMMA core
__device__ __forceinline__ v8f wmma_bf16(v16bf a, v16bf b, v8f c) {
  // (neg_a, A, neg_b, B, c_mod, C, reuse_a, reuse_b)
  return __builtin_amdgcn_wmma_f32_16x16x32_bf16(false, a, false, b, (short)0, c,
                                                 false, false);
}

// A fragment: 16(M) x 32(K) from row-major [rows x stride] LDS.
__device__ __forceinline__ v16bf load_a_frag(const bh16* base, int stride,
                                             int row0, int k0, int lane) {
  const int m  = row0 + (lane & 15);
  const int hi = lane >> 4;
  const bh16* p = base + m * stride + k0 + hi * 8;
  union { int4 i[2]; v16bf v; } u;
  u.i[0] = *(const int4*)(p);
  u.i[1] = *(const int4*)(p + 16);
  return u.v;
}

// B fragment: 32(K) x 16(N), storage N-major: B[k][n] = base[n*stride+k].
__device__ __forceinline__ v16bf load_bt_frag(const bh16* base, int stride,
                                              int k0, int col0, int lane) {
  const int n  = col0 + (lane & 15);
  const int hi = lane >> 4;
  const bh16* p = base + n * stride + k0 + hi * 8;
  union { int4 i[2]; v16bf v; } u;
  u.i[0] = *(const int4*)(p);
  u.i[1] = *(const int4*)(p + 16);
  return u.v;
}

// pack float4 -> 4 bf16, one 8B LDS store
__device__ __forceinline__ void store_bf4(bh16* dst, float4 t) {
  union { bh16 h[4]; uint2 u; } pk;
  pk.h[0] = (bh16)t.x; pk.h[1] = (bh16)t.y;
  pk.h[2] = (bh16)t.z; pk.h[3] = (bh16)t.w;
  *(uint2*)dst = pk.u;
}

__device__ __forceinline__ float gelu_exact(float x) {
  return 0.5f * x * (1.0f + erff(x * 0.7071067811865476f));
}

// ---------------------------------------------------------------- LayerNorm
__launch_bounds__(256)
__global__ void ln_kernel(const float* __restrict__ x, const float* __restrict__ g,
                          const float* __restrict__ bb, bh16* __restrict__ outB,
                          float* __restrict__ outF) {
  __shared__ float2 red[256];
  const int row = blockIdx.x;
  const float* xr = x + (size_t)row * 1024;
  float v[4];
  float s = 0.f, ss = 0.f;
#pragma unroll
  for (int j = 0; j < 4; ++j) {
    v[j] = xr[threadIdx.x + j * 256];
    s += v[j];
    ss += v[j] * v[j];
  }
  red[threadIdx.x] = make_float2(s, ss);
  __syncthreads();
  for (int off = 128; off > 0; off >>= 1) {
    if (threadIdx.x < off) {
      red[threadIdx.x].x += red[threadIdx.x + off].x;
      red[threadIdx.x].y += red[threadIdx.x + off].y;
    }
    __syncthreads();
  }
  const float mean = red[0].x * (1.f / 1024.f);
  const float var  = red[0].y * (1.f / 1024.f) - mean * mean;
  const float inv  = rsqrtf(var + 1e-5f);
#pragma unroll
  for (int j = 0; j < 4; ++j) {
    const int c = threadIdx.x + j * 256;
    const float y = (v[j] - mean) * inv * g[c] + bb[c];
    if (outB) outB[(size_t)row * 1024 + c] = (bh16)y;
    if (outF) outF[(size_t)row * 1024 + c] = y;
  }
}

// ---------------------------------------------------------------- WMMA GEMM
// C[M,N] = A_bf16[M,K] @ W_f32[K,N] + bias.  Block tile 128(M) x 128(N), K-step 64.
// 8 waves as 4x2; wave owns 32 rows x 64 cols = 8 accumulator tiles.
// EPI: 0 = store f32, 1 = f32 + residual, 2 = bf16 gelu(C)
#define GTK 64
#define GLA 72
#define GLB 72

template <int EPI>
__launch_bounds__(256)
__global__ void gemm_wmma(const bh16* __restrict__ A, const float* __restrict__ W,
                          const float* __restrict__ bias, const float* __restrict__ Res,
                          float* __restrict__ Of, bh16* __restrict__ Ob,
                          int M, int N, int K) {
  __shared__ alignas(16) bh16 sA[128 * GLA];   // [m][k]
  __shared__ alignas(16) bh16 sBt[128 * GLB];  // [n][k]  (transposed W tile)
  const int tid  = threadIdx.x;
  const int lane = tid & 31;
  const int w    = tid >> 5;
  const int wr   = w >> 1;     // 0..3 : 32-row group
  const int wc   = w & 1;      // 0..1 : 64-col group
  const int row0 = blockIdx.y * 128;
  const int col0 = blockIdx.x * 128;

  v8f acc[2][4] = {};

  for (int k0 = 0; k0 < K; k0 += GTK) {
    { // A tile 128x64 bf16: thread -> 32 contiguous bf16
      const int r  = tid >> 1;          // 0..127
      const int cc = (tid & 1) * 32;    // 0 or 32
      const bh16* src = A + (size_t)(row0 + r) * K + k0 + cc;
      bh16* dst = &sA[r * GLA + cc];
#if HAVE_ASYNC_LDS
#pragma unroll
      for (int j = 0; j < 4; ++j)
        async_copy_b128(dst + j * 8, src + j * 8);   // 16B per lane, ASYNCcnt
#else
#pragma unroll
      for (int j = 0; j < 4; ++j)
        ((int4*)dst)[j] = ((const int4*)src)[j];
#endif
      if (k0 + GTK < K)
        __builtin_prefetch(src + GTK, 0, 1);
    }
    { // W tile 64x128 f32 -> bf16 transposed: two 4x4 register-transpose blocks
      const int k4 = (tid >> 4) * 4;    // 0..60
      const int n4 = (tid & 15) * 4;    // 0..60
#pragma unroll
      for (int half = 0; half < 2; ++half) {
        const int nn = n4 + half * 64;
        const float* wsrc = W + (size_t)(k0 + k4) * N + col0 + nn;
        const float4 r0 = *(const float4*)(wsrc);
        const float4 r1 = *(const float4*)(wsrc + N);
        const float4 r2 = *(const float4*)(wsrc + 2 * (size_t)N);
        const float4 r3 = *(const float4*)(wsrc + 3 * (size_t)N);
        const float* f0 = &r0.x; const float* f1 = &r1.x;
        const float* f2 = &r2.x; const float* f3 = &r3.x;
#pragma unroll
        for (int j = 0; j < 4; ++j) {
          union { bh16 h[4]; uint2 u; } pk;
          pk.h[0] = (bh16)f0[j]; pk.h[1] = (bh16)f1[j];
          pk.h[2] = (bh16)f2[j]; pk.h[3] = (bh16)f3[j];
          *(uint2*)&sBt[(nn + j) * GLB + k4] = pk.u;
        }
        if (k0 + GTK < K)
          __builtin_prefetch(wsrc + (size_t)GTK * N, 0, 1);
      }
    }
#if HAVE_ASYNC_LDS
    wait_async0();
#endif
    __syncthreads();
#pragma unroll
    for (int kk = 0; kk < GTK; kk += 32) {
      const v16bf a0 = load_a_frag(sA, GLA, wr * 32,      kk, lane);
      const v16bf a1 = load_a_frag(sA, GLA, wr * 32 + 16, kk, lane);
#pragma unroll
      for (int ct = 0; ct < 4; ++ct) {
        const v16bf bfrag = load_bt_frag(sBt, GLB, kk, wc * 64 + ct * 16, lane);
        acc[0][ct] = wmma_bf16(a0, bfrag, acc[0][ct]);
        acc[1][ct] = wmma_bf16(a1, bfrag, acc[1][ct]);
      }
    }
    __syncthreads();
  }

  const int hi = lane >> 4;
  const int nl = lane & 15;
#pragma unroll
  for (int rt = 0; rt < 2; ++rt) {
#pragma unroll
    for (int ct = 0; ct < 4; ++ct) {
#pragma unroll
      for (int r8 = 0; r8 < 8; ++r8) {
        const int mg = row0 + wr * 32 + rt * 16 + r8 + 8 * hi;
        const int ng = col0 + wc * 64 + ct * 16 + nl;
        const float v = acc[rt][ct][r8] + bias[ng];
        const size_t idx = (size_t)mg * N + ng;
        if (EPI == 0)      Of[idx] = v;
        else if (EPI == 1) Of[idx] = v + Res[idx];
        else               Ob[idx] = (bh16)gelu_exact(v);
      }
    }
  }
}

// ---------------------------------------------------------------- Retention
// p: [B*L, 4096] = [q|k|v|xo]; out: ret[B*L, 1024].  grid = (L/64, B*H).
#define RS 72

__launch_bounds__(256)
__global__ void retention_wmma(const float* __restrict__ p, float* __restrict__ ret) {
  __shared__ alignas(16) bh16 Qs[64 * RS];   // [n][hd]
  __shared__ alignas(16) bh16 Ks[64 * RS];   // [m][hd]
  __shared__ alignas(16) bh16 Vt[64 * RS];   // [hd][m]
  __shared__ alignas(16) bh16 Ss[64 * RS];   // [n][m]

  const int b  = blockIdx.y >> 4;
  const int h  = blockIdx.y & 15;
  const int n0 = blockIdx.x * 64;
  const int tid = threadIdx.x, lane = tid & 31, w = tid >> 5;
  const int wr = w >> 1, wc = w & 1;
  const int hi = lane >> 4;

  const float gamma = 1.0f - exp2f(-5.0f - (float)h);
  const float lg    = log2f(gamma);

  { // Q tile [64 x 64] f32 -> bf16 (row-major)
    const int r = tid >> 2;
    const int c = (tid & 3) * 16;
    const float* src = p + ((size_t)(b * 1024 + n0 + r) * 4096) + h * 64 + c;
    bh16* dst = &Qs[r * RS + c];
#pragma unroll
    for (int j = 0; j < 16; j += 4) store_bf4(dst + j, *(const float4*)(src + j));
  }

  v8f acc0 = {}, acc1 = {};

  for (int m0 = 0; m0 <= n0; m0 += 64) {
    { // K tile row-major
      const int r = tid >> 2;
      const int c = (tid & 3) * 16;
      const float* ksrc = p + (size_t)(b * 1024 + m0 + r) * 4096 + 1024 + h * 64 + c;
      bh16* kd = &Ks[r * RS + c];
#pragma unroll
      for (int j = 0; j < 16; j += 4) store_bf4(kd + j, *(const float4*)(ksrc + j));
    }
    { // V tile transposed: 4x4 register transpose
      const int m4 = (tid >> 4) * 4;   // 0..60
      const int d4 = (tid & 15) * 4;   // 0..60
      const float* vsrc = p + (size_t)(b * 1024 + m0 + m4) * 4096 + 2048 + h * 64 + d4;
      const float4 r0 = *(const float4*)(vsrc);
      const float4 r1 = *(const float4*)(vsrc + 4096);
      const float4 r2 = *(const float4*)(vsrc + 8192);
      const float4 r3 = *(const float4*)(vsrc + 12288);
      const float* f0 = &r0.x; const float* f1 = &r1.x;
      const float* f2 = &r2.x; const float* f3 = &r3.x;
#pragma unroll
      for (int j = 0; j < 4; ++j) {
        union { bh16 h4[4]; uint2 u; } pk;
        pk.h4[0] = (bh16)f0[j]; pk.h4[1] = (bh16)f1[j];
        pk.h4[2] = (bh16)f2[j]; pk.h4[3] = (bh16)f3[j];
        *(uint2*)&Vt[(d4 + j) * RS + m4] = pk.u;
      }
    }
    __syncthreads();

    // S[16n x 32m] = Q @ K^T over hd=64   (B[k=hd][n=m] = Ks[m][hd])
    v8f s0 = {}, s1 = {};
#pragma unroll
    for (int kk = 0; kk < 64; kk += 32) {
      const v16bf a  = load_a_frag(Qs, RS, wr * 16, kk, lane);
      const v16bf b0 = load_bt_frag(Ks, RS, kk, wc * 32,      lane);
      const v16bf b1 = load_bt_frag(Ks, RS, kk, wc * 32 + 16, lane);
      s0 = wmma_bf16(a, b0, s0);
      s1 = wmma_bf16(a, b1, s1);
    }

    // causal decay gamma^(n-m) on the C fragment -> Ss (bf16)
#pragma unroll
    for (int r8 = 0; r8 < 8; ++r8) {
      const int nloc = wr * 16 + r8 + 8 * hi;
      const int ng = n0 + nloc;
      {
        const int ml = wc * 32 + (lane & 15);
        const float d = (float)(ng - (m0 + ml));
        Ss[nloc * RS + ml] = (bh16)((d >= 0.f) ? s0[r8] * exp2f(lg * d) : 0.f);
      }
      {
        const int ml = wc * 32 + 16 + (lane & 15);
        const float d = (float)(ng - (m0 + ml));
        Ss[nloc * RS + ml] = (bh16)((d >= 0.f) ? s1[r8] * exp2f(lg * d) : 0.f);
      }
    }
    __syncthreads();

    // ret[16n x 32hd] += S @ V over m=64   (B[k=m][n=hd] = Vt[hd][m])
#pragma unroll
    for (int kk = 0; kk < 64; kk += 32) {
      const v16bf a  = load_a_frag(Ss, RS, wr * 16, kk, lane);
      const v16bf b0 = load_bt_frag(Vt, RS, kk, wc * 32,      lane);
      const v16bf b1 = load_bt_frag(Vt, RS, kk, wc * 32 + 16, lane);
      acc0 = wmma_bf16(a, b0, acc0);
      acc1 = wmma_bf16(a, b1, acc1);
    }
    __syncthreads();
  }

#pragma unroll
  for (int r8 = 0; r8 < 8; ++r8) {
    const int ng = n0 + wr * 16 + r8 + 8 * hi;
    const size_t base = (size_t)(b * 1024 + ng) * 1024 + h * 64;
    ret[base + wc * 32 + (lane & 15)]      = acc0[r8];
    ret[base + wc * 32 + 16 + (lane & 15)] = acc1[r8];
  }
}

// ---------------------------------------------------------------- SiLU gate
__launch_bounds__(256)
__global__ void gate_kernel(const float* __restrict__ ret, const float* __restrict__ p,
                            bh16* __restrict__ outA) {
  const size_t i = (size_t)blockIdx.x * 256 + threadIdx.x;
  const size_t row = i >> 10;
  const size_t col = i & 1023;
  const float xo = p[row * 4096 + 3072 + col];
  const float sg = xo / (1.f + expf(-xo));
  outA[i] = (bh16)(ret[i] * sg);
}

// ---------------------------------------------------------------- Launch
extern "C" void kernel_launch(void* const* d_in, const int* in_sizes, int n_in,
                              void* d_out, int out_size, void* d_ws, size_t ws_size,
                              hipStream_t stream) {
  const float* x      = (const float*)d_in[0];
  const float* proj_w = (const float*)d_in[1];
  const float* proj_b = (const float*)d_in[2];
  const float* reto_w = (const float*)d_in[3];
  const float* reto_b = (const float*)d_in[4];
  const float* ln1_g  = (const float*)d_in[5];
  const float* ln1_b  = (const float*)d_in[6];
  const float* ln2_g  = (const float*)d_in[7];
  const float* ln2_b  = (const float*)d_in[8];
  const float* ffn1_w = (const float*)d_in[9];
  const float* ffn1_b = (const float*)d_in[10];
  const float* ffn2_w = (const float*)d_in[11];
  const float* ffn2_b = (const float*)d_in[12];
  const float* lnf_g  = (const float*)d_in[13];
  const float* lnf_b  = (const float*)d_in[14];

  const size_t MB = 1024ull * 1024ull;
  char* ws = (char*)d_ws;
  bh16*  Abf  = (bh16*)(ws);            //  8 MB bf16 A [4096,1024]
  float* pbuf = (float*)(ws + 8 * MB);  // 64 MB p = [q|k|v|xo]
  bh16*  h1bf = (bh16*)(ws + 8 * MB);   // 32 MB gelu(ffn1) bf16 (reuses pbuf)
  float* retb = (float*)(ws + 72 * MB); // 16 MB retention out
  float* xbuf = (float*)d_out;          // 16 MB running residual in d_out

  const int M = 4096;  // B*L
  (void)hipMemcpyAsync(xbuf, x, (size_t)M * 1024 * sizeof(float),
                       hipMemcpyDeviceToDevice, stream);

  for (int i = 0; i < 2; ++i) {
    const float* pw  = proj_w + (size_t)i * 1024 * 4096;
    const float* pb  = proj_b + (size_t)i * 4096;
    const float* rw  = reto_w + (size_t)i * 1024 * 1024;
    const float* rb  = reto_b + (size_t)i * 1024;
    const float* f1w = ffn1_w + (size_t)i * 1024 * 4096;
    const float* f1b = ffn1_b + (size_t)i * 4096;
    const float* f2w = ffn2_w + (size_t)i * 4096 * 1024;
    const float* f2b = ffn2_b + (size_t)i * 1024;

    ln_kernel<<<M, 256, 0, stream>>>(xbuf, ln1_g + i * 1024, ln1_b + i * 1024,
                                     Abf, nullptr);
    gemm_wmma<0><<<dim3(4096 / 128, M / 128), 256, 0, stream>>>(
        Abf, pw, pb, nullptr, pbuf, nullptr, M, 4096, 1024);
    retention_wmma<<<dim3(1024 / 64, 4 * 16), 256, 0, stream>>>(pbuf, retb);
    gate_kernel<<<(M * 1024) / 256, 256, 0, stream>>>(retb, pbuf, Abf);
    gemm_wmma<1><<<dim3(1024 / 128, M / 128), 256, 0, stream>>>(
        Abf, rw, rb, xbuf, xbuf, nullptr, M, 1024, 1024);
    ln_kernel<<<M, 256, 0, stream>>>(xbuf, ln2_g + i * 1024, ln2_b + i * 1024,
                                     Abf, nullptr);
    gemm_wmma<2><<<dim3(4096 / 128, M / 128), 256, 0, stream>>>(
        Abf, f1w, f1b, nullptr, nullptr, h1bf, M, 4096, 1024);
    gemm_wmma<1><<<dim3(1024 / 128, M / 128), 256, 0, stream>>>(
        h1bf, f2w, f2b, xbuf, xbuf, nullptr, M, 1024, 4096);
  }
  ln_kernel<<<M, 256, 0, stream>>>(xbuf, lnf_g, lnf_b, nullptr, xbuf);
}
